// SelfAttnLSTM_5334349382334
// MI455X (gfx1250) — matrix operand order repaired and checked by reading, not compiled, not run on hardware
//
#include <hip/hip_runtime.h>
#include <hip/hip_bf16.h>

// ---------------------------------------------------------------------------
// SelfAttnLSTM for MI455X (gfx1250, wave32, WMMA).
// All matmuls via v_wmma_f32_16x16x32_bf16 (bf16 inputs, fp32 accumulate).
// B-operand weights pre-swizzled into per-lane WMMA fragment layout; fragment
// loads are contiguous global_load_b128 at constant offsets from a lane base.
// GEMM A-operands staged to LDS with global_load_async_to_lds_b128 (double
// buffered, s_wait_asynccnt).
// ---------------------------------------------------------------------------

typedef __attribute__((ext_vector_type(16))) __bf16 v16bf;
typedef __attribute__((ext_vector_type(8)))  __bf16 v8bf;
typedef __attribute__((ext_vector_type(8)))  float  v8f;

static constexpr int kB    = 16;    // batch
static constexpr int kS    = 512;   // sequence
static constexpr int kDE   = 256;   // msg embed dim
static constexpr int kDEP  = 64;    // pos embed dim
static constexpr int kH    = 512;   // lstm hidden
static constexpr int kG4   = 2048;  // 4*hidden (gates) == D4 feature dim
static constexpr int kDFF  = 2048;
static constexpr int kOUT  = 8;

__device__ __forceinline__ v8f zero8() {
  v8f z = {0.f, 0.f, 0.f, 0.f, 0.f, 0.f, 0.f, 0.f};
  return z;
}

__device__ __forceinline__ v16bf cat8(v8bf lo, v8bf hi) {
  return __builtin_shufflevector(lo, hi, 0, 1, 2, 3, 4, 5, 6, 7, 8, 9, 10, 11,
                                 12, 13, 14, 15);
}

__device__ __forceinline__ v8f wmma_bf16(v16bf a, v16bf b, v8f c) {
  return __builtin_amdgcn_wmma_f32_16x16x32_bf16(false, a, false, b, (short)0, c,
                                                 false, false);
}

// ---- fragment lane-base helpers (contiguous k order, A and B alike) ----------------------
// A 16x32 MxK row-major: lane<16 -> row=lane, K {0..15}; lane>=16 ->
// row=lane-16, K {16..31}. Two 16B runs at +0 and +8 elems.
__device__ __forceinline__ const __bf16* a_lane_base(const __bf16* A, int lda) {
  const int lane = threadIdx.x & 31;
  return A + (size_t)(lane & 15) * lda + 16 * (lane >> 4);
}
__device__ __forceinline__ v16bf load_a_at(const __bf16* lb, size_t eoff) {
  v8bf lo = *(const v8bf*)(lb + eoff);
  v8bf hi = *(const v8bf*)(lb + eoff + 8);
  return cat8(lo, hi);
}
// Packed-B: tile = nt*KT + kt; lane's 16 bf16 contiguous (32B per lane/tile).
__device__ __forceinline__ const __bf16* bp_lane_base(const __bf16* P) {
  return P + (size_t)(threadIdx.x & 31) * 16;
}
// B^T source (row-major T): B[k][n] = T[n][k]; lane's 16 bf16 contiguous.
__device__ __forceinline__ const __bf16* bt_lane_base(const __bf16* T, int ldt) {
  const int lane = threadIdx.x & 31;
  return T + (size_t)(lane & 15) * ldt + 16 * (lane >> 4);
}
__device__ __forceinline__ v16bf load_c16_at(const __bf16* lb, size_t eoff) {
  v8bf lo = *(const v8bf*)(lb + eoff);
  v8bf hi = *(const v8bf*)(lb + eoff + 8);
  return cat8(lo, hi);
}

__device__ __forceinline__ float sigf(float x) {
  return 1.0f / (1.0f + __expf(-x));
}

// ---- gfx1250 async global->LDS copy (ISA 10.x / ASYNCcnt) -----------------
__device__ __forceinline__ void async_g2l_b128(void* lds, const void* gptr) {
  unsigned loff = (unsigned)(size_t)lds;            // LDS aperture: addr[31:0]
  unsigned long long ga = (unsigned long long)(size_t)gptr;
  asm volatile("global_load_async_to_lds_b128 %0, %1, off"
               :: "v"(loff), "v"(ga)
               : "memory");
}
__device__ __forceinline__ void wait_async0() {
  asm volatile("s_wait_asynccnt 0x0" ::: "memory");
}
__device__ __forceinline__ void wait_async1() {
  asm volatile("s_wait_asynccnt 0x1" ::: "memory");
}

// ---------------------------------------------------------------------------
// Elementwise helpers
// ---------------------------------------------------------------------------
__global__ void convert_f32_bf16_kernel(const float* __restrict__ in,
                                        __bf16* __restrict__ out, int n) {
  int i = blockIdx.x * blockDim.x + threadIdx.x;
  if (i < n) out[i] = (__bf16)in[i];
}

__global__ void addvec_kernel(const float* __restrict__ a,
                              const float* __restrict__ b,
                              float* __restrict__ out, int n) {
  int i = blockIdx.x * blockDim.x + threadIdx.x;
  if (i < n) out[i] = a[i] + b[i];
}

__global__ void fill_f32_kernel(float* __restrict__ p, float v, int n) {
  int i = blockIdx.x * blockDim.x + threadIdx.x;
  if (i < n) p[i] = v;
}

__global__ void gather_kernel(const int* __restrict__ tok,
                              const float* __restrict__ table,
                              __bf16* __restrict__ out, int rows, int d) {
  int i = blockIdx.x * blockDim.x + threadIdx.x;
  if (i >= rows * d) return;
  int r = i / d, c = i - r * d;
  out[i] = (__bf16)table[(size_t)tok[r] * d + c];
}

// Pack row-major f32 [K][N] into bf16 WMMA-B fragment layout:
// dst[((nt*KT+kt)*32 + lane)*16 + e] = src[kt*32 + 16*(lane>>4) + e][nt*16 + (lane&15)]
__global__ void pack_b_kernel(const float* __restrict__ src,
                              __bf16* __restrict__ dst, int K, int N) {
  int i = blockIdx.x * blockDim.x + threadIdx.x;
  if (i >= K * N) return;
  const int e    = i & 15;
  const int l    = (i >> 4) & 31;
  const int tile = i >> 9;
  const int KT   = K >> 5;
  const int kt   = tile % KT;
  const int nt   = tile / KT;
  const int k    = kt * 32 + ((l >> 4) << 4) + e;
  const int n    = nt * 16 + (l & 15);
  dst[i] = (__bf16)src[(size_t)k * N + n];
}

// ---------------------------------------------------------------------------
// LSTM recurrence. One workgroup (512 thr = 16 waves) per LSTM direction.
// gates[16,2048] = [x_t | h] @ [Wih ; Whh] via WMMA, fused cell update.
// LDS: h bf16 16x(512+16), c f32 16x512, gates f32 16x2048, bias f32 2048.
// ---------------------------------------------------------------------------
struct LstmCfg {
  const __bf16* x;      // [16][512][Din] bf16 row-major
  const __bf16* wihp;   // packed-B [Din x 2048]
  const __bf16* whhp;   // packed-B [512 x 2048]
  const float*  bias;   // [2048] (bih + bhh)
  int Din;
  int dir;
  int outoff;
};
struct LstmArgs {
  LstmCfg c[4];
  __bf16* feat;         // [16][512][2048]
};

static constexpr int kHS = kH + 16;  // padded bf16 h stride
static constexpr size_t kLstmSmem =
    (size_t)kB * kHS * 2 + (size_t)kB * kH * 4 + (size_t)kB * kG4 * 4 +
    (size_t)kG4 * 4;

__global__ void __launch_bounds__(512) lstm_kernel(LstmArgs args) {
  const LstmCfg cfg = args.c[blockIdx.x];
  extern __shared__ char smem[];
  __bf16* h_sh = (__bf16*)smem;                                   // [16][kHS]
  float*  c_sh = (float*)(smem + (size_t)kB * kHS * 2);           // [16][512]
  float*  g_sh = (float*)(smem + (size_t)kB * kHS * 2 +           // [16][2048]
                          (size_t)kB * kH * 4);
  float*  b_sh = (float*)(smem + (size_t)kB * kHS * 2 +           // [2048]
                          (size_t)kB * kH * 4 + (size_t)kB * kG4 * 4);

  const int tid  = threadIdx.x;
  const int wave = tid >> 5;
  const int lane = tid & 31;
  const int KTx  = cfg.Din >> 5;
  const int KTh  = kH >> 5;  // 16

  for (int i = tid; i < kB * kH; i += 512) {
    c_sh[i] = 0.f;
    h_sh[(i >> 9) * kHS + (i & (kH - 1))] = (__bf16)0.f;
  }
  for (int i = tid; i < kG4; i += 512) b_sh[i] = cfg.bias[i];
  __syncthreads();

  // hoisted lane bases (constant offsets inside the loops)
  const __bf16* xl  = a_lane_base(cfg.x, kS * cfg.Din);
  const __bf16* hl  = a_lane_base(h_sh, kHS);
  const __bf16* wxl = bp_lane_base(cfg.wihp + (size_t)(wave * 8) * KTx * 512);
  const __bf16* whl = bp_lane_base(cfg.whhp + (size_t)(wave * 8) * KTh * 512);

  for (int step = 0; step < kS; ++step) {
    const int t = (cfg.dir > 0) ? step : (kS - 1 - step);

    v8f acc[8];
#pragma unroll
    for (int j = 0; j < 8; ++j) acc[j] = zero8();

    // x_t contribution
    const size_t xoff = (size_t)t * cfg.Din;
    for (int kt = 0; kt < KTx; ++kt) {
      v16bf a = load_a_at(xl, xoff + kt * 32);
#pragma unroll
      for (int j = 0; j < 8; ++j)
        acc[j] = wmma_bf16(a, load_c16_at(wxl, ((size_t)(j * KTx + kt)) * 512),
                           acc[j]);
    }
    // h contribution from LDS
    for (int kt = 0; kt < KTh; ++kt) {
      v16bf a = load_a_at(hl, kt * 32);
      if (kt + 1 < KTh)  // pull next Whh k-tile toward WGP (global_prefetch_b8)
        __builtin_prefetch((const void*)(whl + ((size_t)(kt + 1)) * 512), 0, 0);
#pragma unroll
      for (int j = 0; j < 8; ++j)
        acc[j] = wmma_bf16(a, load_c16_at(whl, ((size_t)(j * KTh + kt)) * 512),
                           acc[j]);
    }

    // Scatter C tiles to gate LDS (ISA C layout).
    const int half = lane >> 4, n = lane & 15;
#pragma unroll
    for (int j = 0; j < 8; ++j) {
      const int n0 = (wave * 8 + j) * 16;
#pragma unroll
      for (int r = 0; r < 8; ++r)
        g_sh[(r + 8 * half) * kG4 + n0 + n] = acc[j][r];
    }
    __syncthreads();

    // Fused cell update
    for (int idx = tid; idx < kB * kH; idx += 512) {
      const int bb = idx >> 9;
      const int jj = idx & (kH - 1);
      const float gi = g_sh[bb * kG4 + jj]          + b_sh[jj];
      const float gf = g_sh[bb * kG4 + kH + jj]     + b_sh[kH + jj];
      const float gg = g_sh[bb * kG4 + 2 * kH + jj] + b_sh[2 * kH + jj];
      const float go = g_sh[bb * kG4 + 3 * kH + jj] + b_sh[3 * kH + jj];
      const float c  = sigf(gf) * c_sh[idx] + sigf(gi) * tanhf(gg);
      const float h  = sigf(go) * tanhf(c);
      c_sh[idx] = c;
      const __bf16 hb = (__bf16)h;
      h_sh[bb * kHS + jj] = hb;
      args.feat[((size_t)bb * kS + t) * kG4 + cfg.outoff + jj] = hb;
    }
    __syncthreads();
  }
}

// ---------------------------------------------------------------------------
// Generic WMMA GEMM: C[M,N] = relu?(A[M,K] @ Bp[K,N] + bias), bf16 in/out.
// A is staged to LDS in 128-wide K chunks with global_load_async_to_lds_b128
// (double buffered, ASYNCcnt); all 8 waves of the block share the staged A.
// Optionally also writes C in packed-B layout (per-512-row batch).
// grid = (M/16, N/512), block = 256. Dynamic LDS = 2*4352 bytes.
// ---------------------------------------------------------------------------
static constexpr int kAStride = 136;                // elems; 272B rows
static constexpr int kABuf    = 16 * kAStride * 2;  // 4352 bytes
static constexpr size_t kGemmSmem = 2 * kABuf;

__device__ __forceinline__ void stage_a_chunk(const __bf16* A, int K, int m0,
                                              int chunk, char* buf) {
  const int i   = threadIdx.x;  // 256 threads -> 16 rows x 16 segments of 16B
  const int m   = i >> 4;
  const int seg = i & 15;
  const __bf16* g = A + (size_t)(m0 + m) * K + chunk * 128 + seg * 8;
  async_g2l_b128(buf + m * (kAStride * 2) + seg * 16, g);
}

__global__ void __launch_bounds__(256)
gemm_bias_kernel(const __bf16* __restrict__ A, const __bf16* __restrict__ Bp,
                 const float* __restrict__ bias, __bf16* __restrict__ C,
                 __bf16* __restrict__ Cp, int M, int N, int K, int relu) {
  extern __shared__ char asmem[];
  char* buf0 = asmem;
  char* buf1 = asmem + kABuf;
  const int wave = threadIdx.x >> 5, lane = threadIdx.x & 31;
  const int m0 = blockIdx.x * 16;
  const int n0 = (blockIdx.y * 8 + wave) * 64;
  const int KT = K >> 5;
  const int NC = K >> 7;  // 128-wide chunks
  const __bf16* bl =
      bp_lane_base(Bp + (size_t)(n0 >> 4) * KT * 512);
  v8f acc[4] = {zero8(), zero8(), zero8(), zero8()};

  stage_a_chunk(A, K, m0, 0, buf0);
  for (int c = 0; c < NC; ++c) {
    char* cur = (c & 1) ? buf1 : buf0;
    char* nxt = (c & 1) ? buf0 : buf1;
    if (c + 1 < NC) {
      stage_a_chunk(A, K, m0, c + 1, nxt);
      wait_async1();
    } else {
      wait_async0();
    }
    __syncthreads();
    const __bf16* al = a_lane_base((const __bf16*)cur, kAStride);
#pragma unroll
    for (int q = 0; q < 4; ++q) {
      const int kt = c * 4 + q;
      v16bf a = load_a_at(al, q * 32);
#pragma unroll
      for (int j = 0; j < 4; ++j)
        acc[j] = wmma_bf16(a, load_c16_at(bl, ((size_t)(j * KT + kt)) * 512),
                           acc[j]);
    }
    __syncthreads();
  }

  const int half = lane >> 4, n = lane & 15;
#pragma unroll
  for (int j = 0; j < 4; ++j) {
#pragma unroll
    for (int r = 0; r < 8; ++r) {
      const int m  = m0 + r + 8 * half;
      const int nn = n0 + j * 16 + n;
      float v = acc[j][r] + bias[nn];
      if (relu) v = fmaxf(v, 0.f);
      const __bf16 bv = (__bf16)v;
      C[(size_t)m * N + nn] = bv;
      if (Cp) {
        const int b = m >> 9;
        const int s = m & 511;
        const int kt2 = s >> 5, rem = s & 31;
        const int lp = ((rem >> 4) << 4) | (nn & 15);
        const int ep = rem & 15;
        const size_t pidx =
            ((((size_t)b * (N >> 4) + (nn >> 4)) * 16 + kt2) * 32 + lp) * 16 +
            ep;
        Cp[pidx] = bv;
      }
    }
  }
}

// ---------------------------------------------------------------------------
// scores[b] = feat2[b] @ feat2[b]^T (512x512, K=2048), fp32 out.
// grid = (32, 1, 16); wave covers 64 cols.
// ---------------------------------------------------------------------------
__global__ void __launch_bounds__(256)
scores_kernel(const __bf16* __restrict__ feat2, float* __restrict__ scores) {
  const int b = blockIdx.z;
  const int wave = threadIdx.x >> 5, lane = threadIdx.x & 31;
  const int m0 = blockIdx.x * 16;
  const int n0 = wave * 64;
  const __bf16* F = feat2 + (size_t)b * kS * kG4;
  const __bf16* al = a_lane_base(F + (size_t)m0 * kG4, kG4);
  const __bf16* btl0 = bt_lane_base(F + (size_t)(n0 + 0)  * kG4, kG4);
  const __bf16* btl1 = bt_lane_base(F + (size_t)(n0 + 16) * kG4, kG4);
  const __bf16* btl2 = bt_lane_base(F + (size_t)(n0 + 32) * kG4, kG4);
  const __bf16* btl3 = bt_lane_base(F + (size_t)(n0 + 48) * kG4, kG4);
  v8f acc[4] = {zero8(), zero8(), zero8(), zero8()};
  for (int k0 = 0; k0 < kG4; k0 += 32) {
    v16bf a = load_a_at(al, k0);
    acc[0] = wmma_bf16(a, load_c16_at(btl0, k0), acc[0]);
    acc[1] = wmma_bf16(a, load_c16_at(btl1, k0), acc[1]);
    acc[2] = wmma_bf16(a, load_c16_at(btl2, k0), acc[2]);
    acc[3] = wmma_bf16(a, load_c16_at(btl3, k0), acc[3]);
  }
  float* out = scores + (size_t)b * kS * kS;
  const int half = lane >> 4, n = lane & 15;
#pragma unroll
  for (int j = 0; j < 4; ++j)
#pragma unroll
    for (int r = 0; r < 8; ++r)
      out[(size_t)(m0 + r + 8 * half) * kS + n0 + j * 16 + n] = acc[j][r];
}

// ---------------------------------------------------------------------------
// Row softmax (512 wide), one wave per row.
// ---------------------------------------------------------------------------
__global__ void __launch_bounds__(256)
softmax_kernel(const float* __restrict__ scores, __bf16* __restrict__ att) {
  const int wave = threadIdx.x >> 5, lane = threadIdx.x & 31;
  const int row = blockIdx.x * 8 + wave;
  const float* src = scores + (size_t)row * kS;
  float vals[16];
  float mx = -3.0e38f;
#pragma unroll
  for (int i = 0; i < 16; ++i) {
    vals[i] = src[lane + i * 32];
    mx = fmaxf(mx, vals[i]);
  }
#pragma unroll
  for (int off = 16; off >= 1; off >>= 1) mx = fmaxf(mx, __shfl_xor(mx, off));
  float sum = 0.f;
#pragma unroll
  for (int i = 0; i < 16; ++i) {
    vals[i] = __expf(vals[i] - mx);
    sum += vals[i];
  }
#pragma unroll
  for (int off = 16; off >= 1; off >>= 1) sum += __shfl_xor(sum, off);
  const float inv = 1.f / sum;
  __bf16* dst = att + (size_t)row * kS;
#pragma unroll
  for (int i = 0; i < 16; ++i) dst[lane + i * 32] = (__bf16)(vals[i] * inv);
}

// ---------------------------------------------------------------------------
// rep[b] = att[b] @ feat2[b] fused with max-pool over s via
// global_atomic_max_num_f32. feat2 consumed in packed-B layout.
// grid = (32, 16, 16); one 16x16 tile per wave, K=512.
// ---------------------------------------------------------------------------
__global__ void __launch_bounds__(256)
rep_maxpool_kernel(const __bf16* __restrict__ att,
                   const __bf16* __restrict__ feat2p,
                   float* __restrict__ pooled) {
  const int b = blockIdx.z;
  const int wave = threadIdx.x >> 5, lane = threadIdx.x & 31;
  const int m0 = blockIdx.x * 16;
  const int n0 = (blockIdx.y * 8 + wave) * 16;
  const int KT = kS >> 5;  // 16
  const __bf16* al = a_lane_base(att + (size_t)b * kS * kS + (size_t)m0 * kS, kS);
  const __bf16* bl = bp_lane_base(feat2p + (size_t)b * kS * kG4 +
                                  (size_t)(n0 >> 4) * KT * 512);
  v8f acc = zero8();
  for (int kt = 0; kt < KT; ++kt)
    acc = wmma_bf16(load_a_at(al, kt * 32), load_c16_at(bl, (size_t)kt * 512),
                    acc);
  float v = acc[0];
#pragma unroll
  for (int r = 1; r < 8; ++r) v = fmaxf(v, acc[r]);
  v = fmaxf(v, __shfl_xor(v, 16));
  if (lane < 16)
    __hip_atomic_fetch_max(&pooled[(size_t)b * kG4 + n0 + lane], v,
                           __ATOMIC_RELAXED, __HIP_MEMORY_SCOPE_AGENT);
}

// ---------------------------------------------------------------------------
// out[16,8] = h1[16,2048] @ ffn_w2[2048,8] + b2
// ---------------------------------------------------------------------------
__global__ void __launch_bounds__(128)
ffn2_kernel(const __bf16* __restrict__ h1, const float* __restrict__ w2,
            const float* __restrict__ b2, float* __restrict__ out) {
  const int tid = threadIdx.x;
  const int b = tid >> 3, o = tid & 7;
  float s = b2[o];
  for (int k = 0; k < kDFF; ++k)
    s += (float)h1[(size_t)b * kDFF + k] * w2[(size_t)k * kOUT + o];
  out[b * kOUT + o] = s;
}

// ---------------------------------------------------------------------------
// Host orchestration
// ---------------------------------------------------------------------------
extern "C" void kernel_launch(void* const* d_in, const int* in_sizes, int n_in,
                              void* d_out, int out_size, void* d_ws,
                              size_t ws_size, hipStream_t stream) {
  const int*   x       = (const int*)d_in[0];
  const int*   pos     = (const int*)d_in[1];
  const float* emb     = (const float*)d_in[2];
  const float* pemb    = (const float*)d_in[3];
  const float* m_wih_f = (const float*)d_in[4];
  const float* m_whh_f = (const float*)d_in[5];
  const float* m_bih_f = (const float*)d_in[6];
  const float* m_bhh_f = (const float*)d_in[7];
  const float* m_wih_b = (const float*)d_in[8];
  const float* m_whh_b = (const float*)d_in[9];
  const float* m_bih_b = (const float*)d_in[10];
  const float* m_bhh_b = (const float*)d_in[11];
  const float* p_wih_f = (const float*)d_in[12];
  const float* p_whh_f = (const float*)d_in[13];
  const float* p_bih_f = (const float*)d_in[14];
  const float* p_bhh_f = (const float*)d_in[15];
  const float* p_wih_b = (const float*)d_in[16];
  const float* p_whh_b = (const float*)d_in[17];
  const float* p_bih_b = (const float*)d_in[18];
  const float* p_bhh_b = (const float*)d_in[19];
  const float* proj_w  = (const float*)d_in[20];
  const float* proj_b  = (const float*)d_in[21];
  const float* ffn_w1  = (const float*)d_in[22];
  const float* ffn_b1  = (const float*)d_in[23];
  const float* ffn_w2  = (const float*)d_in[24];
  const float* ffn_b2  = (const float*)d_in[25];
  float* out = (float*)d_out;

  char* wp = (char*)d_ws;
  auto alloc = [&](size_t bytes) -> char* {
    char* p = wp;
    wp += (bytes + 255) & ~(size_t)255;
    return p;
  };

  __bf16* xemb_m   = (__bf16*)alloc((size_t)kB * kS * kDE * 2);
  __bf16* xemb_p   = (__bf16*)alloc((size_t)kB * kS * kDEP * 2);
  __bf16* w_mwih_f = (__bf16*)alloc((size_t)kDE * kG4 * 2);
  __bf16* w_mwhh_f = (__bf16*)alloc((size_t)kH * kG4 * 2);
  __bf16* w_mwih_b = (__bf16*)alloc((size_t)kDE * kG4 * 2);
  __bf16* w_mwhh_b = (__bf16*)alloc((size_t)kH * kG4 * 2);
  __bf16* w_pwih_f = (__bf16*)alloc((size_t)kDEP * kG4 * 2);
  __bf16* w_pwhh_f = (__bf16*)alloc((size_t)kH * kG4 * 2);
  __bf16* w_pwih_b = (__bf16*)alloc((size_t)kDEP * kG4 * 2);
  __bf16* w_pwhh_b = (__bf16*)alloc((size_t)kH * kG4 * 2);
  __bf16* w_proj   = (__bf16*)alloc((size_t)kG4 * kG4 * 2);
  __bf16* w_ffn1   = (__bf16*)alloc((size_t)kG4 * kDFF * 2);
  float*  bias4    = (float*)alloc((size_t)4 * kG4 * 4);
  __bf16* feat     = (__bf16*)alloc((size_t)kB * kS * kG4 * 2);
  __bf16* feat2    = (__bf16*)alloc((size_t)kB * kS * kG4 * 2);
  __bf16* feat2p   = (__bf16*)alloc((size_t)kB * kS * kG4 * 2);
  float*  scores   = (float*)alloc((size_t)kB * kS * kS * 4);
  __bf16* att      = (__bf16*)alloc((size_t)kB * kS * kS * 2);
  float*  pooled   = (float*)alloc((size_t)kB * kG4 * 4);
  __bf16* pooled_b = (__bf16*)alloc((size_t)kB * kG4 * 2);
  __bf16* h1       = (__bf16*)alloc((size_t)kB * kDFF * 2);

  auto nb = [](size_t n) { return (unsigned)((n + 255) / 256); };
  auto pack = [&](const float* src, __bf16* dst, int K, int N) {
    pack_b_kernel<<<nb((size_t)K * N), 256, 0, stream>>>(src, dst, K, N);
  };

  // 1) weight conversion + swizzle into WMMA-B fragment layout
  pack(m_wih_f, w_mwih_f, kDE, kG4);
  pack(m_whh_f, w_mwhh_f, kH, kG4);
  pack(m_wih_b, w_mwih_b, kDE, kG4);
  pack(m_whh_b, w_mwhh_b, kH, kG4);
  pack(p_wih_f, w_pwih_f, kDEP, kG4);
  pack(p_whh_f, w_pwhh_f, kH, kG4);
  pack(p_wih_b, w_pwih_b, kDEP, kG4);
  pack(p_whh_b, w_pwhh_b, kH, kG4);
  pack(proj_w, w_proj, kG4, kG4);
  pack(ffn_w1, w_ffn1, kG4, kDFF);

  addvec_kernel<<<nb(kG4), 256, 0, stream>>>(m_bih_f, m_bhh_f, bias4 + 0 * kG4, kG4);
  addvec_kernel<<<nb(kG4), 256, 0, stream>>>(m_bih_b, m_bhh_b, bias4 + 1 * kG4, kG4);
  addvec_kernel<<<nb(kG4), 256, 0, stream>>>(p_bih_f, p_bhh_f, bias4 + 2 * kG4, kG4);
  addvec_kernel<<<nb(kG4), 256, 0, stream>>>(p_bih_b, p_bhh_b, bias4 + 3 * kG4, kG4);

  // 2) embedding gathers
  gather_kernel<<<nb((size_t)kB * kS * kDE), 256, 0, stream>>>(
      x, emb, xemb_m, kB * kS, kDE);
  gather_kernel<<<nb((size_t)kB * kS * kDEP), 256, 0, stream>>>(
      pos, pemb, xemb_p, kB * kS, kDEP);

  // 3) four LSTM chains
  LstmArgs la;
  la.feat = feat;
  la.c[0] = {xemb_m, w_mwih_f, w_mwhh_f, bias4 + 0 * kG4, kDE,  +1, 0};
  la.c[1] = {xemb_m, w_mwih_b, w_mwhh_b, bias4 + 1 * kG4, kDE,  -1, 512};
  la.c[2] = {xemb_p, w_pwih_f, w_pwhh_f, bias4 + 2 * kG4, kDEP, +1, 1024};
  la.c[3] = {xemb_p, w_pwih_b, w_pwhh_b, bias4 + 3 * kG4, kDEP, -1, 1536};
  lstm_kernel<<<4, 512, kLstmSmem, stream>>>(la);

  // 4) proj (also emits packed-B copy of feat2)
  gemm_bias_kernel<<<dim3(kB * kS / 16, kG4 / 512), 256, kGemmSmem, stream>>>(
      feat, w_proj, proj_b, feat2, feat2p, kB * kS, kG4, kG4, 0);

  // 5) attention scores + softmax
  scores_kernel<<<dim3(kS / 16, 1, kB), 256, 0, stream>>>(feat2, scores);
  softmax_kernel<<<(kB * kS) / 8, 256, 0, stream>>>(scores, att);

  // 6) rep + fused max-pool
  fill_f32_kernel<<<nb((size_t)kB * kG4), 256, 0, stream>>>(pooled, -3.0e38f,
                                                            kB * kG4);
  rep_maxpool_kernel<<<dim3(kS / 16, kG4 / 128, kB), 256, 0, stream>>>(
      att, feat2p, pooled);

  // 7) FFN
  convert_f32_bf16_kernel<<<nb((size_t)kB * kG4), 256, 0, stream>>>(
      pooled, pooled_b, kB * kG4);
  gemm_bias_kernel<<<dim3(1, kDFF / 512), 256, kGemmSmem, stream>>>(
      pooled_b, w_ffn1, ffn_b1, h1, nullptr, kB, kDFF, kG4, 1);
  ffn2_kernel<<<1, 128, 0, stream>>>(h1, ffn_w2, ffn_b2, out);
}
